// NativeBiMamba_42502996361232
// MI455X (gfx1250) — compile-verified, matrix operand order checked by reading, not run on hardware
//
#include <hip/hip_runtime.h>

// ---------------- problem constants (from reference) ----------------
#define SEQ     2048
#define DMODEL  1024
#define DINNER  2048   // EXPAND * D_MODEL
#define DSTATE  16
#define DTRANK  64
#define DBCW    96     // DT_RANK + 2*D_STATE
#define XZW     4096   // 2 * D_INNER
#define TCHUNK  128    // scan: timesteps of B/C staged in LDS per TDM load

// ---------------- WMMA / TDM types (gfx1250, wave32) ----------------
typedef __attribute__((ext_vector_type(16))) __bf16 v16bf;
typedef __attribute__((ext_vector_type(8)))  __bf16 v8bf;
typedef __attribute__((ext_vector_type(8)))  float  v8f;
typedef __attribute__((ext_vector_type(4)))  unsigned int u32x4;
typedef __attribute__((ext_vector_type(8)))  int i32x8;
typedef __attribute__((ext_vector_type(4)))  int i32x4;

// round-to-nearest-even f32 -> bf16 bits (stored as ushort)
static __device__ __forceinline__ unsigned short f2bf(float f) {
    unsigned u = __float_as_uint(f);
    unsigned r = u + 0x7FFFu + ((u >> 16) & 1u);
    return (unsigned short)(r >> 16);
}

// Load one 16x32 bf16 WMMA operand fragment from a row-major [rows, ld] bf16
// matrix. ISA 16-bit A-matrix layout: lanes 0-15 -> rows with kb=0, lanes
// 16-31 -> same rows with kb=8; per lane elements 0..7 = K kb..kb+7,
// elements 8..15 = K kb+16..kb+23  -> two contiguous 16-byte loads.
static __device__ __forceinline__ v16bf load_frag(const unsigned short* base,
                                                  int ld, int row0, int k0,
                                                  int lane) {
    int r  = row0 + (lane & 15);
    int kb = k0 + ((lane >> 4) << 3);
    const unsigned short* p = base + (size_t)r * ld + kb;
    union U { uint4 u; v8bf h; };
    U lo, hi;
    lo.u = *reinterpret_cast<const uint4*>(p);
    hi.u = *reinterpret_cast<const uint4*>(p + 16);
    v16bf out;
#pragma unroll
    for (int i = 0; i < 8; ++i) { out[i] = lo.h[i]; out[i + 8] = hi.h[i]; }
    return out;
}

// Generic bf16 WMMA GEMM:  C[M,N] = A[M,K] @ B[N,K]^T  (+bias, +softplus)
// One wave per (TM*16)x(TN*16) C tile, 8 waves/block. Epilogue can write f32
// and/or bf16, with column offset and optional row flip (bwd direction).
template <int TM, int TN>
__global__ __launch_bounds__(256) void wmma_gemm_bf16(
    const unsigned short* __restrict__ A, const unsigned short* __restrict__ B,
    const float* __restrict__ bias, float* __restrict__ Cf,
    unsigned short* __restrict__ Cb, int M, int N, int K, int ldc, int col_off,
    int act, int flip) {
    int lane = threadIdx.x & 31;
    int wave = (blockIdx.x * 256 + threadIdx.x) >> 5;
    int ntn  = N / (TN * 16);
    int m0   = (wave / ntn) * (TM * 16);
    int n0   = (wave % ntn) * (TN * 16);

    v8f acc[TM][TN] = {};
    for (int k0 = 0; k0 < K; k0 += 32) {
        // speculative prefetch of the next K tile (global_prefetch_b8)
        __builtin_prefetch(A + (size_t)(m0 + (lane & 15)) * K + k0 + 32, 0, 0);
        __builtin_prefetch(B + (size_t)(n0 + (lane & 15)) * K + k0 + 32, 0, 0);
        v16bf a[TM], b[TN];
#pragma unroll
        for (int i = 0; i < TM; ++i) a[i] = load_frag(A, K, m0 + 16 * i, k0, lane);
#pragma unroll
        for (int j = 0; j < TN; ++j) b[j] = load_frag(B, K, n0 + 16 * j, k0, lane);
#pragma unroll
        for (int i = 0; i < TM; ++i)
#pragma unroll
            for (int j = 0; j < TN; ++j)
                acc[i][j] = __builtin_amdgcn_wmma_f32_16x16x32_bf16(
                    false, a[i], false, b[j], (short)0, acc[i][j], false, false);
    }

    // C/D layout: lane l, vgpr v -> M = v + 8*(l>>4), N = l&15
    int cl = lane & 15;
    int rh = (lane >> 4) << 3;
#pragma unroll
    for (int sm = 0; sm < TM; ++sm)
#pragma unroll
        for (int sn = 0; sn < TN; ++sn)
#pragma unroll
            for (int v = 0; v < 8; ++v) {
                int row = m0 + sm * 16 + rh + v;
                int col = n0 + sn * 16 + cl;
                float val = acc[sm][sn][v];
                if (bias) val += bias[col];
                if (act == 1)  // softplus (fast path)
                    val = (val > 20.0f) ? val : __logf(1.0f + __expf(val));
                int orow = flip ? (M - 1 - row) : row;
                size_t o = (size_t)orow * ldc + col + col_off;
                if (Cf) Cf[o] = val;
                if (Cb) Cb[o] = f2bf(val);
            }
}

// ---------------- elementwise / conversion kernels ----------------
__global__ __launch_bounds__(256) void cvt_bf16_kernel(
    const float* __restrict__ in, unsigned short* __restrict__ out, int n) {
    int i = blockIdx.x * 256 + threadIdx.x;
    if (i < n) out[i] = f2bf(in[i]);
}

__global__ __launch_bounds__(256) void cvt_x_kernel(
    const float* __restrict__ x, unsigned short* __restrict__ out, int flip) {
    int i = blockIdx.x * 256 + threadIdx.x;  // SEQ*DMODEL
    int t = i / DMODEL, c = i % DMODEL;
    int ts = flip ? (SEQ - 1 - t) : t;
    out[i] = f2bf(x[(size_t)ts * DMODEL + c]);
}

// depthwise conv(k=4, left-pad 3) + bias + SiLU over u = xz[:, :DINNER]
__global__ __launch_bounds__(256) void conv_silu_kernel(
    const float* __restrict__ xz, const float* __restrict__ wconv,
    const float* __restrict__ bconv, float* __restrict__ uact,
    unsigned short* __restrict__ ubf) {
    int i = blockIdx.x * 256 + threadIdx.x;  // SEQ*DINNER
    int t = i / DINNER, c = i % DINNER;
    float s = bconv[c];
#pragma unroll
    for (int j = 0; j < 4; ++j) {
        int ts = t - 3 + j;
        if (ts >= 0) s += xz[(size_t)ts * XZW + c] * wconv[c * 4 + j];
    }
    float a = s / (1.f + __expf(-s));
    uact[i] = a;
    ubf[i]  = f2bf(a);
}

__global__ __launch_bounds__(256) void dt_extract_kernel(
    const float* __restrict__ dbc, unsigned short* __restrict__ dtbf) {
    int i = blockIdx.x * 256 + threadIdx.x;  // SEQ*DTRANK
    int t = i >> 6, r = i & 63;
    dtbf[i] = f2bf(dbc[t * DBCW + r]);
}

// Issue one 2D TDM load: dbc[t0 : t0+TCHUNK, 64:96] (f32) -> LDS[lds_off].
// Tile: 32 dwords/row, TCHUNK rows, row stride 96 dwords.  Descriptor per
// CDNA5 ISA: group0 = {flags, lds_addr, global_addr, type=2},
// group1 = {data_size=4B, tensor dims/strides, tile dims}.
static __device__ __forceinline__ void tdm_load_bc(const float* dbc, int t0,
                                                   unsigned lds_off) {
#if __has_builtin(__builtin_amdgcn_tensor_load_to_lds)
    unsigned long long ga =
        (unsigned long long)(dbc + (size_t)t0 * DBCW + DTRANK);
    u32x4 g0;
    g0[0] = 1u;                                   // count=1 valid user desc
    g0[1] = lds_off;                              // lds_addr (bytes)
    g0[2] = (unsigned)(ga & 0xFFFFFFFFu);         // global_addr[31:0]
    g0[3] = (unsigned)((ga >> 32) & 0x01FFFFFFu)  // global_addr[56:32]
            | (2u << 30);                         // type = 2 ("image")
    i32x8 g1;
    g1[0] = (2 << 16);             // data_size = 4 bytes
    g1[1] = (32 << 16);            // tensor_dim0 = 32 (lo16 in bits[31:16])
    g1[2] = (TCHUNK << 16);        // dim0 hi16 = 0 | tensor_dim1 lo16
    g1[3] = (32 << 16);            // dim1 hi16 = 0 | tile_dim0 = 32
    g1[4] = TCHUNK;                // tile_dim1 = TCHUNK, tile_dim2 = 0 (2D)
    g1[5] = DBCW;                  // tensor_dim0_stride[31:0] = 96 dwords
    g1[6] = 0;                     // stride0 hi16 | stride1 lo16
    g1[7] = 0;                     // stride1 hi32
    i32x4 z4 = {};
#if defined(__clang_major__) && __clang_major__ >= 23
    i32x8 z8 = {};
    __builtin_amdgcn_tensor_load_to_lds(g0, g1, z4, z4, z8, 0);
#else
    __builtin_amdgcn_tensor_load_to_lds(g0, g1, z4, z4, 0);
#endif
    __builtin_amdgcn_s_wait_tensorcnt(0);
#else
    (void)dbc; (void)t0; (void)lds_off;
#endif
}

// selective scan: one thread per channel, h[16]/A[16] in VGPRs.  B/C for a
// TCHUNK block of timesteps are staged into LDS by the Tensor Data Mover and
// broadcast-read (same-address ds_load -> conflict free).  Fused +u*D and
// *silu(z) epilogue; writes bf16 y for the following WMMA GEMM.
__global__ __launch_bounds__(256) void scan_kernel(
    const float* __restrict__ delta, const float* __restrict__ uact,
    const float* __restrict__ xz, const float* __restrict__ dbc,
    const float* __restrict__ Alog, const float* __restrict__ Dp,
    unsigned short* __restrict__ ybf) {
    __shared__ float bc[TCHUNK * 32];  // [tt][0:16]=B, [tt][16:32]=C
    int c = blockIdx.x * 256 + threadIdx.x;  // 0..DINNER-1
    float A[DSTATE], h[DSTATE];
#pragma unroll
    for (int n = 0; n < DSTATE; ++n) {
        A[n] = -__expf(Alog[c * DSTATE + n]);
        h[n] = 0.f;
    }
    float dpc = Dp[c];
    for (int t0 = 0; t0 < SEQ; t0 += TCHUNK) {
#if __has_builtin(__builtin_amdgcn_tensor_load_to_lds)
        if (threadIdx.x < 32)  // wave 0 issues the TDM + waits TENSORcnt
            tdm_load_bc(dbc, t0, (unsigned)(unsigned long long)&bc[0]);
#else
        for (int i = threadIdx.x; i < TCHUNK * 32; i += 256) {
            int r = i >> 5, cc = i & 31;
            bc[i] = dbc[(size_t)(t0 + r) * DBCW + DTRANK + cc];
        }
#endif
        __syncthreads();
        for (int tt = 0; tt < TCHUNK; ++tt) {
            int t = t0 + tt;
            float d  = delta[(size_t)t * DINNER + c];
            float u  = uact[(size_t)t * DINNER + c];
            float zz = xz[(size_t)t * XZW + DINNER + c];
            float du = d * u;
            float y  = 0.f;
#pragma unroll
            for (int n = 0; n < DSTATE; ++n) {
                float Bn = bc[tt * 32 + n];
                float Cn = bc[tt * 32 + 16 + n];
                h[n] = __expf(d * A[n]) * h[n] + du * Bn;
                y += h[n] * Cn;
            }
            y += u * dpc;
            y *= zz / (1.f + __expf(-zz));
            ybf[(size_t)t * DINNER + c] = f2bf(y);
        }
        __syncthreads();
    }
}

// ---------------- workspace layout (bytes) ----------------
static const size_t SZ_WINBF  = (size_t)2 * DINNER * DMODEL * 2;  // 8 MB
static const size_t SZ_WXBF   = (size_t)DBCW * DINNER * 2;
static const size_t SZ_WDTBF  = (size_t)DINNER * DTRANK * 2;
static const size_t SZ_WOUTBF = (size_t)DMODEL * DINNER * 2;
static const size_t SZ_WPRJBF = (size_t)DMODEL * 2 * DMODEL * 2;
static const size_t SZ_XBF    = (size_t)SEQ * DMODEL * 2;
static const size_t SZ_XZ     = (size_t)SEQ * XZW * 4;
static const size_t SZ_UACT   = (size_t)SEQ * DINNER * 4;
static const size_t SZ_UBF    = (size_t)SEQ * DINNER * 2;
static const size_t SZ_DBC    = (size_t)SEQ * DBCW * 4;
static const size_t SZ_DTBF   = (size_t)SEQ * DTRANK * 2;
static const size_t SZ_DELTA  = (size_t)SEQ * DINNER * 4;
static const size_t SZ_YBF    = (size_t)SEQ * DINNER * 2;

static const size_t OFF_WINBF  = 0;
static const size_t OFF_WXBF   = OFF_WINBF  + SZ_WINBF;
static const size_t OFF_WDTBF  = OFF_WXBF   + SZ_WXBF;
static const size_t OFF_WOUTBF = OFF_WDTBF  + SZ_WDTBF;
static const size_t OFF_WPRJBF = OFF_WOUTBF + SZ_WOUTBF;
static const size_t OFF_XBF    = OFF_WPRJBF + SZ_WPRJBF;
static const size_t OFF_XZ     = OFF_XBF    + SZ_XBF;
static const size_t OFF_UACT   = OFF_XZ     + SZ_XZ;
static const size_t OFF_UBF    = OFF_UACT   + SZ_UACT;
static const size_t OFF_DBC    = OFF_UBF    + SZ_UBF;
static const size_t OFF_DTBF   = OFF_DBC    + SZ_DBC;
static const size_t OFF_DELTA  = OFF_DTBF   + SZ_DTBF;
static const size_t OFF_YBF    = OFF_DELTA  + SZ_DELTA;
static const size_t OFF_YCAT   = OFF_YBF    + SZ_YBF;  // SEQ x 2*DMODEL bf16

extern "C" void kernel_launch(void* const* d_in, const int* in_sizes, int n_in,
                              void* d_out, int out_size, void* d_ws,
                              size_t ws_size, hipStream_t stream) {
    (void)in_sizes; (void)n_in; (void)out_size; (void)ws_size;
    char* ws = (char*)d_ws;

    unsigned short* winbf  = (unsigned short*)(ws + OFF_WINBF);
    unsigned short* wxbf   = (unsigned short*)(ws + OFF_WXBF);
    unsigned short* wdtbf  = (unsigned short*)(ws + OFF_WDTBF);
    unsigned short* woutbf = (unsigned short*)(ws + OFF_WOUTBF);
    unsigned short* wprjbf = (unsigned short*)(ws + OFF_WPRJBF);
    unsigned short* xbf    = (unsigned short*)(ws + OFF_XBF);
    float*          xz     = (float*)(ws + OFF_XZ);
    float*          uact   = (float*)(ws + OFF_UACT);
    unsigned short* ubf    = (unsigned short*)(ws + OFF_UBF);
    float*          dbc    = (float*)(ws + OFF_DBC);
    unsigned short* dtbf   = (unsigned short*)(ws + OFF_DTBF);
    float*          deltab = (float*)(ws + OFF_DELTA);
    unsigned short* ybf    = (unsigned short*)(ws + OFF_YBF);
    unsigned short* ycat   = (unsigned short*)(ws + OFF_YCAT);

    const float* x     = (const float*)d_in[0];
    const float* wproj = (const float*)d_in[19];
    const float* bproj = (const float*)d_in[20];
    float* out = (float*)d_out;

    auto cvt = [&](const float* in, unsigned short* o, int n) {
        cvt_bf16_kernel<<<(n + 255) / 256, 256, 0, stream>>>(in, o, n);
    };
    // 64x64 wave tile (16 WMMA per K-step, 8 fragment loads)
    auto gemm44 = [&](const unsigned short* A, const unsigned short* B,
                      const float* bias, float* Cf, unsigned short* Cb, int M,
                      int N, int K, int ldc, int col_off, int act, int flip) {
        int waves = (M / 64) * (N / 64);
        wmma_gemm_bf16<4, 4><<<waves / 8, 256, 0, stream>>>(
            A, B, bias, Cf, Cb, M, N, K, ldc, col_off, act, flip);
    };
    // 32x32 wave tile for the narrow N=96 GEMM
    auto gemm22 = [&](const unsigned short* A, const unsigned short* B,
                      const float* bias, float* Cf, unsigned short* Cb, int M,
                      int N, int K, int ldc, int col_off, int act, int flip) {
        int waves = (M / 32) * (N / 32);
        wmma_gemm_bf16<2, 2><<<waves / 8, 256, 0, stream>>>(
            A, B, bias, Cf, Cb, M, N, K, ldc, col_off, act, flip);
    };

    cvt(wproj, wprjbf, DMODEL * 2 * DMODEL);  // final projection weight (once)

    for (int dir = 0; dir < 2; ++dir) {
        int o = 1 + dir * 9;
        const float* win   = (const float*)d_in[o + 0];
        const float* wconv = (const float*)d_in[o + 1];
        const float* bconv = (const float*)d_in[o + 2];
        const float* wx    = (const float*)d_in[o + 3];
        const float* wdt   = (const float*)d_in[o + 4];
        const float* bdt   = (const float*)d_in[o + 5];
        const float* Alog  = (const float*)d_in[o + 6];
        const float* Dp    = (const float*)d_in[o + 7];
        const float* wout  = (const float*)d_in[o + 8];

        // convert weights + (flipped) input to bf16
        cvt(win, winbf, 2 * DINNER * DMODEL);
        cvt(wx, wxbf, DBCW * DINNER);
        cvt(wdt, wdtbf, DINNER * DTRANK);
        cvt(wout, woutbf, DMODEL * DINNER);
        cvt_x_kernel<<<(SEQ * DMODEL) / 256, 256, 0, stream>>>(x, xbf, dir);

        // xz = x @ win.T   [SEQ, 4096] f32
        gemm44(xbf, winbf, nullptr, xz, nullptr, SEQ, 2 * DINNER, DMODEL,
               2 * DINNER, 0, 0, 0);
        // depthwise conv + SiLU -> uact (f32) + ubf (bf16)
        conv_silu_kernel<<<(SEQ * DINNER) / 256, 256, 0, stream>>>(
            xz, wconv, bconv, uact, ubf);
        // dbc = u @ wx.T   [SEQ, 96]
        gemm22(ubf, wxbf, nullptr, dbc, nullptr, SEQ, DBCW, DINNER, DBCW, 0, 0,
               0);
        // dt slice -> bf16
        dt_extract_kernel<<<(SEQ * DTRANK) / 256, 256, 0, stream>>>(dbc, dtbf);
        // delta = softplus(dt @ wdt.T + bdt)   [SEQ, 2048]
        gemm44(dtbf, wdtbf, bdt, deltab, nullptr, SEQ, DINNER, DTRANK, DINNER,
               0, 1, 0);
        // selective scan (TDM-staged B/C in LDS) -> y (bf16)
        scan_kernel<<<DINNER / 256, 256, 0, stream>>>(deltab, uact, xz, dbc,
                                                      Alog, Dp, ybf);
        // y @ wout.T -> ycat[:, dir*1024 : +1024], rows flipped for bwd
        gemm44(ybf, woutbf, nullptr, nullptr, ycat, SEQ, DMODEL, DINNER,
               2 * DMODEL, dir * DMODEL, 0, dir);
    }

    // out = ycat @ wproj.T + bproj   [SEQ, 1024] f32
    gemm44(ycat, wprjbf, bproj, out, nullptr, SEQ, DMODEL, 2 * DMODEL, DMODEL,
           0, 0, 0);
}
